// SIM2_38603166056700
// MI455X (gfx1250) — compile-verified, matrix operand order checked by reading, not run on hardware
//
#include <hip/hip_runtime.h>
#include <stdint.h>

typedef __attribute__((ext_vector_type(16))) __bf16   v16bf;
typedef __attribute__((ext_vector_type(8)))  __bf16   v8bf;
typedef __attribute__((ext_vector_type(8)))  float    v8f;
typedef __attribute__((ext_vector_type(4)))  float    v4f;
typedef __attribute__((ext_vector_type(4)))  uint32_t v4u;

#define GAS __attribute__((address_space(1)))

#define N_ROWS 8192
#define DIM    4096

// ---------------------------------------------------------------------------
// f32 -> bf16 (RNE) conversion, 8 elements / thread, 16B global stores
// ---------------------------------------------------------------------------
__global__ __launch_bounds__(256)
void f32_to_bf16_kernel(const float* __restrict__ src,
                        uint16_t* __restrict__ dst, long long n) {
    long long i = ((long long)blockIdx.x * blockDim.x + threadIdx.x) * 8;
    if (i >= n) return;
    const GAS v4f* s = (const GAS v4f*)(src + i);
    v4f a = s[0];
    v4f b = s[1];
    auto cvt = [](float x) -> uint32_t {
        uint32_t u = __float_as_uint(x);
        return (u + 0x7FFFu + ((u >> 16) & 1u)) >> 16;   // round-to-nearest-even
    };
    v4u o;
    o.x = cvt(a.x) | (cvt(a.y) << 16);
    o.y = cvt(a.z) | (cvt(a.w) << 16);
    o.z = cvt(b.x) | (cvt(b.y) << 16);
    o.w = cvt(b.z) | (cvt(b.w) << 16);
    *(GAS v4u*)(dst + i) = o;
}

// ---------------------------------------------------------------------------
// F = A(bf16, N_ROWS x DIM) * B(bf16, DIM x DIM, row = output channel)^T + bias
// Workgroup: 256 threads (8 waves), tile 128(M) x 128(N).
// Wave grid 2(M) x 4(N): each wave computes 64x32 = 4x2 WMMA 16x16 tiles.
// v_wmma_f32_16x16x32_bf16, f32 accumulate. Global AS1 addressing:
// uniform 64-bit base + 32-bit per-lane offsets -> global_load_b128 (SADDR).
// ---------------------------------------------------------------------------
__global__ __launch_bounds__(256)
void gemm_bf16_wmma(const uint16_t* __restrict__ A,
                    const uint16_t* __restrict__ B,
                    const float* __restrict__ bias,
                    float* __restrict__ F) {
    const int lane  = threadIdx.x & 31;
    const int wave  = threadIdx.x >> 5;
    const int waveM = wave >> 2;   // 0..1
    const int waveN = wave & 3;    // 0..3
    const int m0 = blockIdx.x * 128 + waveM * 64;
    const int n0 = blockIdx.y * 128 + waveN * 32;
    const int l15   = lane & 15;
    const int lhi   = lane >> 4;        // 0 or 1
    const int khalf = lhi * 8;          // ISA 16-bit A/B layout: upper half-wave holds K+8 / K+24

    const GAS uint16_t* Ag = (const GAS uint16_t*)A;
    const GAS uint16_t* Bg = (const GAS uint16_t*)B;

    // 32-bit element offsets (max 33.5M elements -> fits u32)
    uint32_t aoff[4], boff[2];
#pragma unroll
    for (int mi = 0; mi < 4; ++mi)
        aoff[mi] = (uint32_t)(m0 + mi * 16 + l15) * DIM + khalf;
#pragma unroll
    for (int ni = 0; ni < 2; ++ni)
        boff[ni] = (uint32_t)(n0 + ni * 16 + l15) * DIM + khalf;

    v8f acc[4][2] = {};

    union Frag { v16bf v; v8bf h[2]; };

    for (int k = 0; k < DIM; k += 32) {
        Frag a[4], b[2];
#pragma unroll
        for (int mi = 0; mi < 4; ++mi) {
            a[mi].h[0] = *(const GAS v8bf*)(Ag + aoff[mi] + k);       // K 0..7   (+khalf)
            a[mi].h[1] = *(const GAS v8bf*)(Ag + aoff[mi] + k + 16);  // K 16..23 (+khalf)
        }
#pragma unroll
        for (int ni = 0; ni < 2; ++ni) {
            b[ni].h[0] = *(const GAS v8bf*)(Bg + boff[ni] + k);
            b[ni].h[1] = *(const GAS v8bf*)(Bg + boff[ni] + k + 16);
        }
        if (k + 288 <= DIM) {   // lookahead prefetch, 8 k-tiles ahead (WGP scope)
#pragma unroll
            for (int mi = 0; mi < 4; ++mi)
                __builtin_prefetch((const uint16_t*)(Ag + aoff[mi] + k + 256), 0, 3);
#pragma unroll
            for (int ni = 0; ni < 2; ++ni)
                __builtin_prefetch((const uint16_t*)(Bg + boff[ni] + k + 256), 0, 3);
        }

#pragma unroll
        for (int mi = 0; mi < 4; ++mi)
#pragma unroll
            for (int ni = 0; ni < 2; ++ni)
                acc[mi][ni] = __builtin_amdgcn_wmma_f32_16x16x32_bf16(
                    /*neg_a=*/false, a[mi].v,
                    /*neg_b=*/false, b[ni].v,
                    /*c_mod=*/(short)0, acc[mi][ni],
                    /*reuse_a=*/false, /*reuse_b=*/false);
    }

    // Epilogue: C/D layout -> lane l, vgpr r holds (M = r + 8*(l>>4), N = l&15).
    // F is written once, read once by softmax -> stream with non-temporal stores
    // so fb/wb stay L2-resident.
    const GAS float* biasg = (const GAS float*)bias;
    GAS float* Fg = (GAS float*)F;
#pragma unroll
    for (int ni = 0; ni < 2; ++ni) {
        const int col = n0 + ni * 16 + l15;
        const float bv = biasg[col];
#pragma unroll
        for (int mi = 0; mi < 4; ++mi) {
            const uint32_t rbase = (uint32_t)(m0 + mi * 16 + lhi * 8);
#pragma unroll
            for (int r = 0; r < 8; ++r)
                __builtin_nontemporal_store(acc[mi][ni][r] + bv,
                                            Fg + (rbase + r) * (uint32_t)DIM + col);
        }
    }
}

// ---------------------------------------------------------------------------
// One block per row: p = softmax(F[row]); out = (accumulate ? out : f) + p
// 256 threads x 16 contiguous cols each; LDS tree reductions.
// ---------------------------------------------------------------------------
__global__ __launch_bounds__(256)
void softmax_add_kernel(const float* __restrict__ Fm,
                        const float* __restrict__ f,
                        float* __restrict__ out, int accumulate) {
    __shared__ float red[256];
    const int tid = threadIdx.x;
    const size_t base = (size_t)blockIdx.x * DIM;
    const int c0 = tid * 16;

    v4f v[4];
    const GAS v4f* src = (const GAS v4f*)(Fm + base + c0);
    float lmax = -3.402823466e38f;
#pragma unroll
    for (int i = 0; i < 4; ++i) {
        v[i] = __builtin_nontemporal_load(src + i);   // F read exactly once
        lmax = fmaxf(lmax, fmaxf(fmaxf(v[i].x, v[i].y), fmaxf(v[i].z, v[i].w)));
    }
    red[tid] = lmax;
    __syncthreads();
#pragma unroll
    for (int s = 128; s > 0; s >>= 1) {
        if (tid < s) red[tid] = fmaxf(red[tid], red[tid + s]);
        __syncthreads();
    }
    const float rmax = red[0];
    __syncthreads();

    float lsum = 0.0f;
#pragma unroll
    for (int i = 0; i < 4; ++i) {
        v[i].x = __expf(v[i].x - rmax);
        v[i].y = __expf(v[i].y - rmax);
        v[i].z = __expf(v[i].z - rmax);
        v[i].w = __expf(v[i].w - rmax);
        lsum += v[i].x + v[i].y + v[i].z + v[i].w;
    }
    red[tid] = lsum;
    __syncthreads();
#pragma unroll
    for (int s = 128; s > 0; s >>= 1) {
        if (tid < s) red[tid] += red[tid + s];
        __syncthreads();
    }
    const float inv = 1.0f / red[0];

    GAS v4f* o = (GAS v4f*)(out + base + c0);
    const GAS v4f* fin = (const GAS v4f*)(f + base + c0);
#pragma unroll
    for (int i = 0; i < 4; ++i) {
        v4f bb = accumulate ? o[i] : fin[i];
        o[i] = bb + v[i] * inv;
    }
}

// ---------------------------------------------------------------------------
// Host-side orchestration (all on `stream`, deterministic, ws-only scratch)
// ws layout: fb (N*D bf16, 64 MiB) | wb (D*D bf16, 32 MiB) | F (N*D f32, 128 MiB)
// ---------------------------------------------------------------------------
extern "C" void kernel_launch(void* const* d_in, const int* in_sizes, int n_in,
                              void* d_out, int out_size, void* d_ws, size_t ws_size,
                              hipStream_t stream) {
    const float* f  = (const float*)d_in[0];
    const float* W1 = (const float*)d_in[1];
    const float* b1 = (const float*)d_in[2];
    const float* W2 = (const float*)d_in[3];
    const float* b2 = (const float*)d_in[4];
    float* out = (float*)d_out;

    char* ws = (char*)d_ws;
    uint16_t* fb = (uint16_t*)ws;
    uint16_t* wb = (uint16_t*)(ws + (size_t)N_ROWS * DIM * 2);
    float*    Fm = (float*)(ws + (size_t)N_ROWS * DIM * 2 + (size_t)DIM * DIM * 2);

    const long long nf = (long long)N_ROWS * DIM;   // 33,554,432
    const long long nw = (long long)DIM * DIM;      // 16,777,216
    const int cvtBlk = 256, cvtPerThread = 8;

    dim3 ggrid(N_ROWS / 128, DIM / 128);

    // convert f once; W1 -> gemm -> softmax(out = f + p1); W2 -> gemm -> softmax(out += p2)
    f32_to_bf16_kernel<<<(int)(nf / (cvtBlk * cvtPerThread)), cvtBlk, 0, stream>>>(f,  fb, nf);
    f32_to_bf16_kernel<<<(int)(nw / (cvtBlk * cvtPerThread)), cvtBlk, 0, stream>>>(W1, wb, nw);
    gemm_bf16_wmma<<<ggrid, 256, 0, stream>>>(fb, wb, b1, Fm);
    softmax_add_kernel<<<N_ROWS, 256, 0, stream>>>(Fm, f, out, 0);

    f32_to_bf16_kernel<<<(int)(nw / (cvtBlk * cvtPerThread)), cvtBlk, 0, stream>>>(W2, wb, nw);
    gemm_bf16_wmma<<<ggrid, 256, 0, stream>>>(fb, wb, b2, Fm);
    softmax_add_kernel<<<N_ROWS, 256, 0, stream>>>(Fm, f, out, 1);
}
